// clique_MPNN_13778255086001
// MI455X (gfx1250) — matrix operand order repaired and checked by reading, not compiled, not run on hardware
//
#include <hip/hip_runtime.h>
#include <hip/hip_bf16.h>

#define NNODES  100000
#define NGROUP  50
#define NPG     2000
#define NEDGE   1600000
#define HDIM    256
#define HMID    32
#define LRES    3
#define LRELU_S 0.01f
#define BN_EPS_S 1e-5f
#define PENALTY_S 4.0f
#define INV_SQRT_NPG 0.022360679774997897f   // rsqrt(2000)

typedef __bf16 bf16_t;
typedef bf16_t v8bf  __attribute__((ext_vector_type(8)));
typedef bf16_t v16bf __attribute__((ext_vector_type(16)));
typedef float  v8f   __attribute__((ext_vector_type(8)));
typedef unsigned int u32x4 __attribute__((ext_vector_type(4)));
typedef int          i32x4 __attribute__((ext_vector_type(4)));
typedef int          i32x8 __attribute__((ext_vector_type(8)));

__device__ __forceinline__ float leaky_f(float x) { return x > 0.f ? x : LRELU_S * x; }

// ---------------------------------------------------------------- mask / agg
__global__ void k_mask_init(const float* __restrict__ x, int* __restrict__ m0, int* __restrict__ m1) {
    int i = blockIdx.x * 256 + threadIdx.x;
    if (i >= NNODES) return;
    int a = (fabsf(x[i]) > 0.f) ? 1 : 0;
    m0[i] = a; m1[i] = a;
}

__global__ void k_mask_copy(const int* __restrict__ cur, int* __restrict__ nxt) {
    int i = blockIdx.x * 256 + threadIdx.x;
    if (i >= NNODES) return;
    nxt[i] = cur[i];
}

__global__ void k_mask_prop(const int* __restrict__ cur, int* __restrict__ nxt,
                            const int* __restrict__ src, const int* __restrict__ dst) {
    int e = blockIdx.x * 256 + threadIdx.x;
    if (e >= NEDGE) return;
    if (cur[src[e]]) atomicMax(&nxt[dst[e]], 1);
}

__global__ void k_agg_scalar(const float* __restrict__ x, const int* __restrict__ src,
                             const int* __restrict__ dst, float* __restrict__ agg0) {
    int e = blockIdx.x * 256 + threadIdx.x;
    if (e >= NEDGE) return;
    atomicAdd(&agg0[dst[e]], x[src[e]]);
}

// one block per edge, one channel per thread
__global__ void k_agg_vec(const float* __restrict__ h, const int* __restrict__ src,
                          const int* __restrict__ dst, float* __restrict__ agg) {
    int e = blockIdx.x;
    int c = threadIdx.x;
    atomicAdd(&agg[(size_t)dst[e] * HDIM + c], h[(size_t)src[e] * HDIM + c]);
}

// ---------------------------------------------------------------- layer-0 front: z0*W1+b1, relu -> bf16
__global__ void k_front0(const float* __restrict__ x, const float* __restrict__ agg0,
                         const float* __restrict__ eps0, const float* __restrict__ W1,
                         const float* __restrict__ b1, bf16_t* __restrict__ y1) {
    size_t idx = (size_t)blockIdx.x * 256 + threadIdx.x;
    if (idx >= (size_t)NNODES * HDIM) return;
    int i = (int)(idx >> 8), c = (int)(idx & 255);
    float z0 = (1.f + eps0[0]) * x[i] + agg0[i];
    float y  = fmaxf(fmaf(z0, W1[c], b1[c]), 0.f);
    y1[idx] = (bf16_t)y;
}

// ---------------------------------------------------------------- z-prep: bf16((1+eps)*h + agg)
__global__ void k_zprep(const float* __restrict__ h, const float* __restrict__ agg,
                        const float* __restrict__ epsArr, int li, bf16_t* __restrict__ zb) {
    size_t idx = (size_t)blockIdx.x * 256 + threadIdx.x;
    if (idx >= (size_t)NNODES * HDIM) return;
    float e1 = 1.f + epsArr[li];
    zb[idx] = (bf16_t)fmaf(e1, h[idx], agg[idx]);
}

// ---------------------------------------------------------------- weight pack to B-fragment order (ct-major)
// packed[ct*4096 + kt*512 + lane*16 + i] = W[kt*32 + (lane>>4)*8 + (i&7) + (i>>3)*16][ct*16 + (lane&15)]
__global__ void k_pack_w(const float* __restrict__ W, bf16_t* __restrict__ out) {
    int idx = blockIdx.x * 256 + threadIdx.x;           // 65536 total
    int i  = idx & 15;
    int l  = (idx >> 4) & 31;
    int kt = (idx >> 9) & 7;
    int ct = idx >> 12;
    int k = kt * 32 + (l >> 4) * 8 + (i & 7) + (i >> 3) * 16;
    int n = ct * 16 + (l & 15);
    out[idx] = (bf16_t)W[k * HDIM + n];
}

// ---------------------------------------------------------------- WMMA GEMM with TDM-staged weights
// C = relu(A(N,256) @ W(256,256) + b); W staged to LDS in two 64KB halves by the Tensor Data Mover
template <bool OUT_BF16>
__global__ void k_gemm_relu(const bf16_t* __restrict__ A, const bf16_t* __restrict__ Wp,
                            const float* __restrict__ bias, float* __restrict__ outF,
                            bf16_t* __restrict__ outB) {
    __shared__ bf16_t smW[32768];                       // 64KB: 8 col-tiles x 8 k-tiles
    int wave = threadIdx.x >> 5;
    int lane = threadIdx.x & 31;
    int rt = blockIdx.x * 8 + wave;
    bool active = (rt < NNODES / 16);

    // A-fragment: row = rt*16 + (lane&15), two 8-elem runs at k0=(lane>>4)*8 and k0+16
    v16bf afrag[8];
    if (active) {
        const bf16_t* arow = A + (size_t)(rt * 16 + (lane & 15)) * HDIM + ((lane >> 4) * 8);
#pragma unroll
        for (int kt = 0; kt < 8; kt++) {
            v8bf lo = *reinterpret_cast<const v8bf*>(arow + kt * 32);
            v8bf hi = *reinterpret_cast<const v8bf*>(arow + kt * 32 + 16);
            v16bf a;
#pragma unroll
            for (int j = 0; j < 8; j++) { a[j] = lo[j]; a[j + 8] = hi[j]; }
            afrag[kt] = a;
        }
    }
    int row0 = rt * 16 + (lane >> 4) * 8;

    for (int stage = 0; stage < 2; stage++) {
        // ---- Tensor Data Mover: DMA one 64KB contiguous weight half into LDS (wave 0 only)
        if (threadIdx.x < 32) {
            unsigned long long ga =
                (unsigned long long)(const void*)(Wp + (size_t)stage * 32768);
            unsigned int ldsOff = (unsigned int)(unsigned long long)(uintptr_t)&smW[0];
            u32x4 g0;
            g0.x = 1u;                                          // count=1, user descriptor
            g0.y = ldsOff;                                      // lds_addr (bytes)
            g0.z = (unsigned int)(ga & 0xffffffffu);            // global_addr[31:0]
            g0.w = (unsigned int)((ga >> 32) & 0x01ffffffu)     // global_addr[56:32]
                 | (2u << 30);                                  // type = 2 ("image")
            i32x8 g1;
            g1.s0 = 0x00020000;                                 // data_size = 4B, no multicast
            g1.s1 = (int)(16384u << 16);                        // tensor_dim0[15:0] = 16384 dwords
            g1.s2 = 0x00010000;                                 // tensor_dim0[31:16]=0, tensor_dim1=1
            g1.s3 = (int)(16384u << 16);                        // tile_dim0 = 16384
            g1.s4 = 1;                                          // tile_dim1 = 1
            g1.s5 = 16384;                                      // tensor_dim0_stride
            g1.s6 = 0; g1.s7 = 0;
            i32x4 gz4 = {0, 0, 0, 0};
            i32x8 gz8 = {0, 0, 0, 0, 0, 0, 0, 0};
            __builtin_amdgcn_tensor_load_to_lds(g0, g1, gz4, gz4, gz8, 0);
            __builtin_amdgcn_s_wait_tensorcnt(0);
        }
        __syncthreads();

        if (active) {
            for (int cl = 0; cl < 8; cl++) {
                v8f c = {0.f, 0.f, 0.f, 0.f, 0.f, 0.f, 0.f, 0.f};
                const bf16_t* wb = &smW[cl * 4096 + lane * 16];
#pragma unroll
                for (int kt = 0; kt < 8; kt++) {
                    v16bf b = *reinterpret_cast<const v16bf*>(wb + kt * 512);
                    c = __builtin_amdgcn_wmma_f32_16x16x32_bf16(false, afrag[kt], false, b,
                                                                (short)0, c, false, false);
                }
                int ct = stage * 8 + cl;
                int col = ct * 16 + (lane & 15);
                float bv = bias[col];
#pragma unroll
                for (int v = 0; v < 8; v++) {
                    float xv = fmaxf(c[v] + bv, 0.f);
                    size_t o = (size_t)(row0 + v) * HDIM + col;
                    if (OUT_BF16) outB[o] = (bf16_t)xv;
                    else          outF[o] = xv;
                }
            }
        }
        __syncthreads();   // protect smW before next stage's DMA overwrites it
    }
}

// ---------------------------------------------------------------- batch-norm stats / finalize / apply
__global__ void k_bn_stats(const float* __restrict__ z, float* __restrict__ sum,
                           float* __restrict__ sumsq) {
    int c = threadIdx.x;
    int r0 = blockIdx.x * 256;
    int rend = r0 + 256; if (rend > NNODES) rend = NNODES;
    float s = 0.f, q = 0.f;
    for (int r = r0; r < rend; r++) {
        float v = z[(size_t)r * HDIM + c];
        s += v; q += v * v;
    }
    atomicAdd(&sum[c], s);
    atomicAdd(&sumsq[c], q);
}

__global__ void k_bn_finalize(const float* __restrict__ sum, const float* __restrict__ sumsq,
                              const float* __restrict__ g, const float* __restrict__ b,
                              float* __restrict__ scale, float* __restrict__ shift) {
    int c = threadIdx.x;
    float m = sum[c] * (1.f / NNODES);
    float var = sumsq[c] * (1.f / NNODES) - m * m;
    float sc = g[c] * rsqrtf(var + BN_EPS_S);
    scale[c] = sc;
    shift[c] = b[c] - m * sc;
}

// fused: leaky(bn(z2)) [+h] -> *mask*inv_sqrt_n -> htmp, plus 2nd-BN partial stats
__global__ void k_combine(const float* __restrict__ z2, const float* __restrict__ hOld,
                          const int* __restrict__ mask, const float* __restrict__ scale,
                          const float* __restrict__ shift, float* __restrict__ htmp,
                          float* __restrict__ sum2, float* __restrict__ sumsq2, int residual) {
    int c = threadIdx.x;
    int r0 = blockIdx.x * 256;
    int rend = r0 + 256; if (rend > NNODES) rend = NNODES;
    float sc = scale[c], sh = shift[c];
    float s = 0.f, q = 0.f;
    for (int r = r0; r < rend; r++) {
        size_t o = (size_t)r * HDIM + c;
        float t = leaky_f(fmaf(z2[o], sc, sh));
        if (residual) t += hOld[o];
        t *= mask[r] ? INV_SQRT_NPG : 0.f;
        htmp[o] = t;
        s += t; q += t * t;
    }
    atomicAdd(&sum2[c], s);
    atomicAdd(&sumsq2[c], q);
}

__global__ void k_bn_apply(const float* __restrict__ htmp, const float* __restrict__ scale,
                           const float* __restrict__ shift, float* __restrict__ h) {
    size_t idx = (size_t)blockIdx.x * 256 + threadIdx.x;
    if (idx >= (size_t)NNODES * HDIM) return;
    int c = (int)(idx & 255);
    h[idx] = fmaf(htmp[idx], scale[c], shift[c]);
}

// ---------------------------------------------------------------- head: lin1 (256->32) leaky mask, lin2 (32->1) leaky mask
__global__ void k_head(const float* __restrict__ h, const float* __restrict__ W1,
                       const float* __restrict__ b1, const float* __restrict__ W2,
                       const float* __restrict__ b2, const int* __restrict__ mask,
                       float* __restrict__ hfin) {
    __shared__ float sh[8 * HDIM];
    int wave = threadIdx.x >> 5;
    int lane = threadIdx.x & 31;
    int node0 = blockIdx.x * 8;
    for (int t = threadIdx.x; t < 8 * HDIM; t += 256) {
        int ni = node0 + (t >> 8);
        sh[t] = (ni < NNODES) ? h[(size_t)ni * HDIM + (t & 255)] : 0.f;
    }
    __syncthreads();
    int node = node0 + wave;
    if (node >= NNODES) return;
    const float* hv = &sh[wave * HDIM];
    float acc = 0.f;
    for (int k = 0; k < HDIM; k++) acc = fmaf(hv[k], W1[k * HMID + lane], acc);
    float mk = mask[node] ? 1.f : 0.f;
    float t1 = leaky_f(acc + b1[lane]) * mk;
    float p = t1 * W2[lane];
    for (int off = 16; off > 0; off >>= 1) p += __shfl_down(p, off, 32);
    if (lane == 0) hfin[node] = leaky_f(p + b2[0]) * mk;
}

// ---------------------------------------------------------------- per-group reductions (groups contiguous)
__global__ void k_gminmax(const float* __restrict__ hfin, float* __restrict__ bmax,
                          float* __restrict__ bmin) {
    __shared__ float smax[256], smin[256];
    int g = blockIdx.x, t = threadIdx.x;
    float mx = -INFINITY, mn = INFINITY;
    for (int i = t; i < NPG; i += 256) {
        float v = hfin[g * NPG + i];
        mx = fmaxf(mx, v); mn = fminf(mn, v);
    }
    smax[t] = mx; smin[t] = mn; __syncthreads();
    for (int s = 128; s > 0; s >>= 1) {
        if (t < s) { smax[t] = fmaxf(smax[t], smax[t + s]); smin[t] = fminf(smin[t], smin[t + s]); }
        __syncthreads();
    }
    if (t == 0) { bmax[g] = smax[0]; bmin[g] = smin[0]; }
}

__global__ void k_probs(const float* __restrict__ hfin, const float* __restrict__ bmax,
                        const float* __restrict__ bmin, float* __restrict__ outP,
                        float* __restrict__ sG, float* __restrict__ ssqG) {
    __shared__ float ss[256], sq[256];
    int g = blockIdx.x, t = threadIdx.x;
    float bmn = bmin[g];
    float inv = 1.f / (bmax[g] + 1e-6f - bmn);
    float s = 0.f, q = 0.f;
    for (int i = t; i < NPG; i += 256) {
        int n = g * NPG + i;
        float p = (hfin[n] - bmn) * inv;
        outP[n] = p;
        s += p; q += p * p;
    }
    ss[t] = s; sq[t] = q; __syncthreads();
    for (int st = 128; st > 0; st >>= 1) {
        if (t < st) { ss[t] += ss[t + st]; sq[t] += sq[t + st]; }
        __syncthreads();
    }
    if (t == 0) { sG[g] = ss[0]; ssqG[g] = sq[0]; }
}

__global__ void k_edge_weight(const float* __restrict__ probs, const int* __restrict__ src,
                              const int* __restrict__ dst, float* __restrict__ ewG) {
    __shared__ float bins[64];
    int t = threadIdx.x;
    if (t < 64) bins[t] = 0.f;
    __syncthreads();
    int e = blockIdx.x * 256 + t;
    if (e < NEDGE) {
        int s0 = src[e], d0 = dst[e];
        atomicAdd(&bins[s0 / NPG], probs[s0] * probs[d0]);
    }
    __syncthreads();
    if (t < NGROUP) atomicAdd(&ewG[t], bins[t]);
}

__global__ void k_loss(const float* __restrict__ sG, const float* __restrict__ ssqG,
                       const float* __restrict__ ewG, float* __restrict__ outL) {
    int g = threadIdx.x;
    if (g >= NGROUP) return;
    float s = sG[g];
    float pair = 0.5f * s * s;
    float ew = 0.5f * ewG[g];
    float dist = (pair - ssqG[g]) - ew;
    outL[g] = PENALTY_S * dist * 0.5f - 0.5f * ew;
}

// ================================================================ launcher
extern "C" void kernel_launch(void* const* d_in, const int* in_sizes, int n_in,
                              void* d_out, int out_size, void* d_ws, size_t ws_size,
                              hipStream_t stream) {
    (void)in_sizes; (void)n_in; (void)out_size; (void)ws_size;
    const float* x    = (const float*)d_in[0];
    const int*   ei   = (const int*)d_in[1];
    const int*   src  = ei;
    const int*   dst  = ei + NEDGE;
    const float* eps0 = (const float*)d_in[3];
    const float* c0W1 = (const float*)d_in[4];
    const float* c0b1 = (const float*)d_in[5];
    const float* c0W2 = (const float*)d_in[6];
    const float* c0b2 = (const float*)d_in[7];
    const float* c0g  = (const float*)d_in[8];
    const float* c0bb = (const float*)d_in[9];
    const float* epsL = (const float*)d_in[10];
    const float* cW1  = (const float*)d_in[11];
    const float* cb1  = (const float*)d_in[12];
    const float* cW2  = (const float*)d_in[13];
    const float* cb2  = (const float*)d_in[14];
    const float* cg   = (const float*)d_in[15];
    const float* cbb  = (const float*)d_in[16];
    const float* bn1g = (const float*)d_in[17];
    const float* bn1b = (const float*)d_in[18];
    const float* bnsg = (const float*)d_in[19];
    const float* bnsb = (const float*)d_in[20];
    const float* l1W  = (const float*)d_in[21];
    const float* l1b  = (const float*)d_in[22];
    const float* l2W  = (const float*)d_in[23];
    const float* l2b  = (const float*)d_in[24];

    float* outP = (float*)d_out;
    float* outL = outP + NNODES;

    // ---------------- workspace carve-up
    char* base = (char*)d_ws;
    size_t off = 0;
    auto carve = [&](size_t bytes) -> char* {
        char* p = base + off;
        off = (off + bytes + 255) & ~(size_t)255;
        return p;
    };
    const size_t NH4 = (size_t)NNODES * HDIM * 4;
    float*  h     = (float*)carve(NH4);
    float*  agg   = (float*)carve(NH4);          // also reused as htmp
    float*  htmp  = agg;
    char*   zreg  = carve(NH4);                  // zbf16 (first half) then z2 f32
    bf16_t* zbf   = (bf16_t*)zreg;
    float*  z2    = (float*)zreg;
    bf16_t* y1    = (bf16_t*)carve((size_t)NNODES * HDIM * 2);
    float*  agg0  = (float*)carve((size_t)NNODES * 4);
    int*    m0    = (int*)carve((size_t)NNODES * 4);
    int*    m1    = (int*)carve((size_t)NNODES * 4);
    float*  hfin  = (float*)carve((size_t)NNODES * 4);
    bf16_t* pw[7];
    for (int i = 0; i < 7; i++) pw[i] = (bf16_t*)carve(65536 * 2);  // c0W2, cW1[0..2], cW2[0..2]
    float* stats  = (float*)carve(8 * 256 * 4);
    float* sum    = stats;            float* sumsq  = stats + 256;
    float* scale  = stats + 512;      float* shift  = stats + 768;
    float* sum2   = stats + 1024;     float* sumsq2 = stats + 1280;
    float* scale2 = stats + 1536;     float* shift2 = stats + 1792;
    float* grp    = (float*)carve(5 * 64 * 4);
    float* bmax   = grp;       float* bmin = grp + 64;
    float* sG     = grp + 128; float* ssqG = grp + 192; float* ewG = grp + 256;

    const int TB = 256;
    const int NBLK_NODE = (NNODES + TB - 1) / TB;               // 391
    const int NBLK_EDGE = NEDGE / TB;                           // 6250
    const int NBLK_NH   = (int)(((size_t)NNODES * HDIM) / TB);  // 100000
    const int NBLK_GEMM = (NNODES / 16 + 7) / 8;                // 782

    // ---------------- pack weights (bf16, B-fragment order, ct-major for TDM staging)
    k_pack_w<<<256, TB, 0, stream>>>(c0W2, pw[0]);
    for (int l = 0; l < LRES; l++) {
        k_pack_w<<<256, TB, 0, stream>>>(cW1 + (size_t)l * HDIM * HDIM, pw[1 + l]);
        k_pack_w<<<256, TB, 0, stream>>>(cW2 + (size_t)l * HDIM * HDIM, pw[4 + l]);
    }

    // ---------------- initial mask + layer 0
    k_mask_init<<<NBLK_NODE, TB, 0, stream>>>(x, m0, m1);
    k_mask_prop<<<NBLK_EDGE, TB, 0, stream>>>(m0, m1, src, dst);   // cur mask = m1

    (void)hipMemsetAsync(agg0, 0, (size_t)NNODES * 4, stream);
    k_agg_scalar<<<NBLK_EDGE, TB, 0, stream>>>(x, src, dst, agg0);
    k_front0<<<NBLK_NH, TB, 0, stream>>>(x, agg0, eps0, c0W1, c0b1, y1);
    k_gemm_relu<false><<<NBLK_GEMM, TB, 0, stream>>>(y1, pw[0], c0b2, z2, nullptr);

    (void)hipMemsetAsync(sum, 0, 2 * 256 * 4, stream);
    k_bn_stats<<<NBLK_NODE, TB, 0, stream>>>(z2, sum, sumsq);
    k_bn_finalize<<<1, 256, 0, stream>>>(sum, sumsq, c0g, c0bb, scale, shift);
    (void)hipMemsetAsync(sum2, 0, 2 * 256 * 4, stream);
    k_combine<<<NBLK_NODE, TB, 0, stream>>>(z2, h, m1, scale, shift, htmp, sum2, sumsq2, 0);
    k_bn_finalize<<<1, 256, 0, stream>>>(sum2, sumsq2, bn1g, bn1b, scale2, shift2);
    k_bn_apply<<<NBLK_NH, TB, 0, stream>>>(htmp, scale2, shift2, h);

    // ---------------- residual GIN layers
    int* mcur = m1;
    int* mnxt = m0;
    for (int l = 0; l < LRES; l++) {
        k_mask_copy<<<NBLK_NODE, TB, 0, stream>>>(mcur, mnxt);
        k_mask_prop<<<NBLK_EDGE, TB, 0, stream>>>(mcur, mnxt, src, dst);

        (void)hipMemsetAsync(agg, 0, NH4, stream);
        k_agg_vec<<<NEDGE, TB, 0, stream>>>(h, src, dst, agg);
        k_zprep<<<NBLK_NH, TB, 0, stream>>>(h, agg, epsL, l, zbf);

        k_gemm_relu<true><<<NBLK_GEMM, TB, 0, stream>>>(zbf, pw[1 + l], cb1 + l * HDIM, nullptr, y1);
        k_gemm_relu<false><<<NBLK_GEMM, TB, 0, stream>>>(y1, pw[4 + l], cb2 + l * HDIM, z2, nullptr);

        (void)hipMemsetAsync(sum, 0, 2 * 256 * 4, stream);
        k_bn_stats<<<NBLK_NODE, TB, 0, stream>>>(z2, sum, sumsq);
        k_bn_finalize<<<1, 256, 0, stream>>>(sum, sumsq, cg + l * HDIM, cbb + l * HDIM, scale, shift);
        (void)hipMemsetAsync(sum2, 0, 2 * 256 * 4, stream);
        k_combine<<<NBLK_NODE, TB, 0, stream>>>(z2, h, mnxt, scale, shift, htmp, sum2, sumsq2, 1);
        k_bn_finalize<<<1, 256, 0, stream>>>(sum2, sumsq2, bnsg + l * HDIM, bnsb + l * HDIM, scale2, shift2);
        k_bn_apply<<<NBLK_NH, TB, 0, stream>>>(htmp, scale2, shift2, h);

        int* tmp = mcur; mcur = mnxt; mnxt = tmp;
    }

    // ---------------- head + per-group loss
    k_head<<<NNODES / 8, TB, 0, stream>>>(h, l1W, l1b, l2W, l2b, mcur, hfin);
    k_gminmax<<<NGROUP, TB, 0, stream>>>(hfin, bmax, bmin);
    k_probs<<<NGROUP, TB, 0, stream>>>(hfin, bmax, bmin, outP, sG, ssqG);
    (void)hipMemsetAsync(ewG, 0, 64 * 4, stream);
    k_edge_weight<<<NBLK_EDGE, TB, 0, stream>>>(outP, src, dst, ewG);
    k_loss<<<1, 64, 0, stream>>>(sG, ssqG, ewG, outL);
}
